// DGCNN_42683384988202
// MI455X (gfx1250) — matrix operand order repaired and checked by reading, not compiled, not run on hardware
//
#include <hip/hip_runtime.h>
#include <math.h>

// ---------------------------------------------------------------------------
// DGCNN forward for MI455X (gfx1250, wave32, WMMA).
// All GEMM-shaped work (pairwise Gram matrices for kNN, edge-MLP, h5
// projection) runs on v_wmma_f32_16x16x32_bf16 with f32 accumulation.
// Features live in bf16 padded to multiples of 32 channels; all tensors fit
// in the 192MB L2 so the design minimizes passes, not HBM traffic.
//
// Input flattening assumption (setup_inputs() dict insertion order,
// recursive): x, then per edge layer l=0..3: W, gamma, beta, mean, var,
// then W5, bn5(g,b,m,v), W6, bn6(...), W7, b7, bn7(...), W8, b8.
// ---------------------------------------------------------------------------

#define B_    8
#define N_    2048
#define KNN_  20
#define EPS_  1e-5f

typedef __attribute__((ext_vector_type(16))) __bf16 bf16x16;
typedef __attribute__((ext_vector_type(8)))  __bf16 bf16x8;
typedef __attribute__((ext_vector_type(8)))  float  f32x8;

__device__ __forceinline__ f32x8 wmma_bf16(bf16x16 a, bf16x16 b, f32x8 c) {
  // (neg_a, A, neg_b, B, c_mod, C, reuse_a, reuse_b)
  return __builtin_amdgcn_wmma_f32_16x16x32_bf16(false, a, false, b, (short)0, c,
                                                 false, false);
}

// 16-bit A-matrix 16x32 fragment (ISA 7.12.2): lane m=lane%16;
// lanes 0-15 hold K = kb*32 + {0..7, 16..23}; lanes 16-31 K = kb*32 + {8..15, 24..31}
__device__ __forceinline__ bf16x16 ld_afrag(const __bf16* rowBase, int kb, int lane) {
  const int half8 = (lane & 16) ? 8 : 0;
  const __bf16* p0 = rowBase + kb * 32 + half8;
  union { bf16x16 v; bf16x8 h[2]; } u;
  u.h[0] = *reinterpret_cast<const bf16x8*>(p0);
  u.h[1] = *reinterpret_cast<const bf16x8*>(p0 + 16);
  return u.v;
}

// 16-bit B-matrix 32x16 fragment: col = lane%16; lanes 0-15 hold K=kb*32+{0..7, 16..23},
// lanes 16-31 hold K=kb*32+{8..15, 24..31} (same K order as A).
__device__ __forceinline__ bf16x16 ld_bfrag(const __bf16* colBase, int kb, int lane) {
  const __bf16* p0 = colBase + kb * 32 + ((lane & 16) ? 8 : 0);
  union { bf16x16 v; bf16x8 h[2]; } u;
  u.h[0] = *reinterpret_cast<const bf16x8*>(p0);
  u.h[1] = *reinterpret_cast<const bf16x8*>(p0 + 16);
  return u.v;
}

__device__ __forceinline__ float lrelu(float y) { return (y >= 0.f) ? y : 0.2f * y; }

// ---------------------------------------------------------------------------
// prep_xt: x (8,3,2048) f32 -> xt (8*2048, 32) bf16, channels 3..31 zero
// ---------------------------------------------------------------------------
__global__ void prep_xt(const float* __restrict__ x, __bf16* __restrict__ xt) {
  int p = blockIdx.x * blockDim.x + threadIdx.x;
  if (p >= B_ * N_) return;
  int b = p / N_, n = p % N_;
  const float* xp = x + (size_t)b * 3 * N_ + n;
  __bf16* o = xt + (size_t)p * 32;
#pragma unroll
  for (int c = 0; c < 32; ++c) o[c] = (__bf16)0.f;
  o[0] = (__bf16)xp[0];
  o[1] = (__bf16)xp[N_];
  o[2] = (__bf16)xp[2 * N_];
}

// ---------------------------------------------------------------------------
// prep_weights: fold BN into per-channel scale/shift, convert/zero-pad W->bf16
// Wb[o][c'] layout: [diff part (cinHalf used of Cpad) | ctr part (cinHalf of Cpad)]
// ---------------------------------------------------------------------------
__global__ void prep_weights(const float* __restrict__ W,
                             const float* __restrict__ gamma,
                             const float* __restrict__ beta,
                             const float* __restrict__ mean,
                             const float* __restrict__ var,
                             __bf16* __restrict__ Wb,
                             float* __restrict__ scale, float* __restrict__ shift,
                             int cout, int K2, int Cpad, int cinHalf, int cinW) {
  int id = blockIdx.x * blockDim.x + threadIdx.x;
  if (id < cout) {
    float sc = gamma[id] / sqrtf(var[id] + EPS_);
    scale[id] = sc;
    shift[id] = beta[id] - mean[id] * sc;
  }
  if (id < cout * K2) {
    int o = id / K2, c = id % K2;
    float v = 0.f;
    if (c < cinHalf) v = W[(size_t)o * cinW + c];
    else if (c >= Cpad && c < Cpad + cinHalf) v = W[(size_t)o * cinW + cinHalf + (c - Cpad)];
    Wb[id] = (__bf16)v;
  }
}

// ---------------------------------------------------------------------------
// compute_sq: squared L2 norm per point (over padded channels; pads are 0)
// ---------------------------------------------------------------------------
template <int CPAD>
__global__ void compute_sq(const __bf16* __restrict__ feat, int stride,
                           float* __restrict__ sq) {
  int p = blockIdx.x * blockDim.x + threadIdx.x;
  if (p >= B_ * N_) return;
  const __bf16* f = feat + (size_t)p * stride;
  float s = 0.f;
#pragma unroll
  for (int c = 0; c < CPAD; ++c) { float v = (float)f[c]; s += v * v; }
  sq[p] = s;
}

// ---------------------------------------------------------------------------
// knn_kernel: one wave per 16-row tile. WMMA Gram sweep over all 2048 cols,
// pd = 2*inner - |r|^2 - |c|^2, per-lane sorted top-20 (row split over lane
// halves), final LDS merge -> knn_idx (local col indices).
// ---------------------------------------------------------------------------
template <int CPAD>
__global__ void __launch_bounds__(32) knn_kernel(const __bf16* __restrict__ feat,
                                                 int stride,
                                                 const float* __restrict__ sq,
                                                 int* __restrict__ knn_idx) {
  constexpr int NKB = CPAD / 32;
  const int tile = blockIdx.x;            // [0, B*N/16)
  const int b = tile / (N_ / 16);
  const int row0 = tile * 16;             // global row base
  const int lane = threadIdx.x;
  const int m = lane & 15;
  const int half8 = (lane & 16) ? 8 : 0;

  // Persistent A fragments for the 16 rows of this tile.
  bf16x16 afrag[NKB];
  const __bf16* arow = feat + (size_t)(row0 + m) * stride;
#pragma unroll
  for (int kb = 0; kb < NKB; ++kb) afrag[kb] = ld_afrag(arow, kb, lane);

  float rsq[8];
#pragma unroll
  for (int v = 0; v < 8; ++v) rsq[v] = sq[row0 + v + half8];

  __shared__ float tl[16][17];
  __shared__ float mv[32][KNN_];
  __shared__ int   mi[32][KNN_];

  float tv[KNN_];
  int   ti[KNN_];
#pragma unroll
  for (int j = 0; j < KNN_; ++j) { tv[j] = -3.0e38f; ti[j] = 0; }

  const int r = lane & 15;
  const int hsel = (lane >> 4) & 1;
  const float* sqb = sq + b * N_;
  const __bf16* featb = feat + (size_t)b * N_ * stride;

  for (int ct = 0; ct < N_ / 16; ++ct) {
    f32x8 acc = {0.f, 0.f, 0.f, 0.f, 0.f, 0.f, 0.f, 0.f};
    const int colbase = ct * 16;
    const __bf16* colptr = featb + (size_t)(colbase + m) * stride;
#pragma unroll
    for (int kb = 0; kb < NKB; ++kb) {
      bf16x16 bfrag = ld_bfrag(colptr, kb, lane);
      acc = wmma_bf16(afrag[kb], bfrag, acc);
    }
    float csq = sqb[colbase + m];
    __syncthreads();
#pragma unroll
    for (int v = 0; v < 8; ++v) {
      tl[v + half8][m] = 2.f * acc[v] - rsq[v] - csq;
    }
    __syncthreads();
    // lane handles row r, candidate cols colbase + hsel*8 + j
#pragma unroll
    for (int j = 0; j < 8; ++j) {
      float val = tl[r][hsel * 8 + j];
      int id = colbase + hsel * 8 + j;
      if (val > tv[KNN_ - 1]) {
        tv[KNN_ - 1] = val; ti[KNN_ - 1] = id;
#pragma unroll
        for (int q = KNN_ - 1; q >= 1; --q) {
          if (tv[q] > tv[q - 1]) {
            float tf = tv[q]; tv[q] = tv[q - 1]; tv[q - 1] = tf;
            int   tq = ti[q]; ti[q] = ti[q - 1]; ti[q - 1] = tq;
          }
        }
      }
    }
    __syncthreads();
  }

#pragma unroll
  for (int j = 0; j < KNN_; ++j) { mv[lane][j] = tv[j]; mi[lane][j] = ti[j]; }
  __syncthreads();
  if (lane < 16) {
    int ia = 0, ib = 0;
    int* outp = knn_idx + (size_t)(row0 + lane) * KNN_;
#pragma unroll
    for (int j = 0; j < KNN_; ++j) {
      float va = mv[lane][ia], vb = mv[lane + 16][ib];
      if (va >= vb) { outp[j] = mi[lane][ia]; ++ia; }
      else          { outp[j] = mi[lane + 16][ib]; ++ib; }
    }
  }
}

// ---------------------------------------------------------------------------
// edge_conv: one block (128 thr / 4 waves) per point. Builds 32 x K2 bf16
// edge-feature tile [nbr-ctr | ctr] in LDS (rows 20..31 zero), WMMA against
// BN-folded weights, fused scale/shift + LReLU, max over k, bf16 out slice.
// ---------------------------------------------------------------------------
template <int CPAD, int CINH, int K2, int COUT>
__global__ void __launch_bounds__(128) edge_conv(const __bf16* __restrict__ featIn,
                                                 int strideIn,
                                                 const int* __restrict__ knn_idx,
                                                 const __bf16* __restrict__ Wb,
                                                 const float* __restrict__ scale,
                                                 const float* __restrict__ shift,
                                                 __bf16* __restrict__ outFeat,
                                                 int strideOut) {
  constexpr int NKB = K2 / 32;
  constexpr int NCT = COUT / 16;
  const int p = blockIdx.x;               // global point index
  const int b = p / N_;
  const int tid = threadIdx.x;

  __shared__ __bf16 ef[32][K2];
  __shared__ int    sidx[KNN_];
  __shared__ float  ctrf[CPAD];

  if (tid < KNN_) sidx[tid] = knn_idx[(size_t)p * KNN_ + tid];
  const __bf16* ctrp = featIn + (size_t)p * strideIn;
  for (int c = tid; c < CPAD; c += 128) ctrf[c] = (float)ctrp[c];
  __syncthreads();

  for (int e = tid; e < 32 * K2; e += 128) {
    int rr = e / K2, c = e % K2;
    float v = 0.f;
    if (rr < KNN_) {
      if (c < CPAD) {
        const __bf16* nb = featIn + ((size_t)b * N_ + sidx[rr]) * (size_t)strideIn;
        v = (float)nb[c] - ctrf[c];
      } else {
        v = ctrf[c - CPAD];
      }
    }
    ef[rr][c] = (__bf16)v;
  }
  __syncthreads();

  const int wave = tid >> 5, lane = tid & 31;
  const int mcol = lane & 15;
  const int half8 = (lane & 16) ? 8 : 0;

  for (int ct = wave; ct < NCT; ct += 4) {
    f32x8 acc0 = {0.f, 0.f, 0.f, 0.f, 0.f, 0.f, 0.f, 0.f};
    f32x8 acc1 = {0.f, 0.f, 0.f, 0.f, 0.f, 0.f, 0.f, 0.f};
    const int o = ct * 16 + mcol;
    const __bf16* wrow = Wb + (size_t)o * K2;
#pragma unroll
    for (int kb = 0; kb < NKB; ++kb) {
      bf16x16 bfrag = ld_bfrag(wrow, kb, lane);
      bf16x16 a0 = ld_afrag(&ef[mcol][0], kb, lane);
      bf16x16 a1 = ld_afrag(&ef[16 + mcol][0], kb, lane);
      acc0 = wmma_bf16(a0, bfrag, acc0);
      acc1 = wmma_bf16(a1, bfrag, acc1);
    }
    float sc = scale[o], sh = shift[o];
    float mx = -3.0e38f;
#pragma unroll
    for (int v = 0; v < 8; ++v) {        // rows 0..15: always valid (< 20)
      mx = fmaxf(mx, lrelu(acc0[v] * sc + sh));
    }
#pragma unroll
    for (int v = 0; v < 8; ++v) {        // rows 16..31: valid only if < 20
      int m1 = 16 + v + half8;
      if (m1 < KNN_) mx = fmaxf(mx, lrelu(acc1[v] * sc + sh));
    }
    mx = fmaxf(mx, __shfl_xor(mx, 16, 32));  // merge row-halves (lane, lane+16)
    if (lane < 16) outFeat[(size_t)p * strideOut + o] = (__bf16)mx;
  }
}

// ---------------------------------------------------------------------------
// h5_gemm: h5 = lrelu(bn(hcat @ W5^T)); M=B*N, K=512, Nout=256.
// Block = 128 thr / 4 waves, 16 rows x 256 cols per block, f32 output.
// ---------------------------------------------------------------------------
__global__ void __launch_bounds__(128) h5_gemm(const __bf16* __restrict__ hcat,
                                               const __bf16* __restrict__ W5b,
                                               const float* __restrict__ scale,
                                               const float* __restrict__ shift,
                                               float* __restrict__ h5) {
  const int row0 = blockIdx.x * 16;
  const int wave = threadIdx.x >> 5, lane = threadIdx.x & 31;
  const int mcol = lane & 15;
  const int half8 = (lane & 16) ? 8 : 0;
  const __bf16* arow = hcat + (size_t)(row0 + mcol) * 512;

  f32x8 acc[4];
#pragma unroll
  for (int t = 0; t < 4; ++t) acc[t] = (f32x8){0.f, 0.f, 0.f, 0.f, 0.f, 0.f, 0.f, 0.f};

  for (int kb = 0; kb < 16; ++kb) {
    bf16x16 a = ld_afrag(arow, kb, lane);
#pragma unroll
    for (int t = 0; t < 4; ++t) {
      int o = (wave * 4 + t) * 16 + mcol;
      bf16x16 bf = ld_bfrag(W5b + (size_t)o * 512, kb, lane);
      acc[t] = wmma_bf16(a, bf, acc[t]);
    }
  }
#pragma unroll
  for (int t = 0; t < 4; ++t) {
    int o = (wave * 4 + t) * 16 + mcol;
    float sc = scale[o], sh = shift[o];
#pragma unroll
    for (int v = 0; v < 8; ++v) {
      h5[(size_t)(row0 + v + half8) * 256 + o] = lrelu(acc[t][v] * sc + sh);
    }
  }
}

// ---------------------------------------------------------------------------
// pool: per batch, max and mean over N of h5 -> g0 (8, 512) = [max | mean]
// ---------------------------------------------------------------------------
__global__ void __launch_bounds__(256) pool_kernel(const float* __restrict__ h5,
                                                   float* __restrict__ g0) {
  int b = blockIdx.x;
  int o = threadIdx.x;
  const float* p = h5 + (size_t)b * N_ * 256 + o;
  float mx = -3.0e38f, sm = 0.f;
  for (int n = 0; n < N_; ++n) {
    float v = p[(size_t)n * 256];
    mx = fmaxf(mx, v);
    sm += v;
  }
  g0[b * 512 + o] = mx;
  g0[b * 512 + 256 + o] = sm * (1.f / N_);
}

// ---------------------------------------------------------------------------
// head: tiny FC stack (8x512 -> 512 -> 256 -> 3) in one block, f32 VALU.
// ---------------------------------------------------------------------------
__global__ void __launch_bounds__(256) head_kernel(
    const float* __restrict__ g0,
    const float* __restrict__ W6, const float* __restrict__ g6,
    const float* __restrict__ be6, const float* __restrict__ me6,
    const float* __restrict__ va6,
    const float* __restrict__ W7, const float* __restrict__ b7,
    const float* __restrict__ g7, const float* __restrict__ be7,
    const float* __restrict__ me7, const float* __restrict__ va7,
    const float* __restrict__ W8, const float* __restrict__ b8,
    float* __restrict__ out) {
  __shared__ float gbuf[512], h6[512], h7[256];
  int t = threadIdx.x;
  for (int b = 0; b < B_; ++b) {
    gbuf[t] = g0[b * 512 + t];
    gbuf[t + 256] = g0[b * 512 + 256 + t];
    __syncthreads();
    for (int oo = t; oo < 512; oo += 256) {
      const float* w = W6 + (size_t)oo * 512;
      float s = 0.f;
      for (int c = 0; c < 512; ++c) s += gbuf[c] * w[c];
      float sc = g6[oo] / sqrtf(va6[oo] + EPS_);
      h6[oo] = lrelu((s - me6[oo]) * sc + be6[oo]);
    }
    __syncthreads();
    {
      const float* w = W7 + (size_t)t * 512;
      float s = 0.f;
      for (int c = 0; c < 512; ++c) s += h6[c] * w[c];
      s += b7[t];
      float sc = g7[t] / sqrtf(va7[t] + EPS_);
      h7[t] = lrelu((s - me7[t]) * sc + be7[t]);
    }
    __syncthreads();
    if (t < 3) {
      const float* w = W8 + t * 256;
      float s = 0.f;
      for (int c = 0; c < 256; ++c) s += h7[c] * w[c];
      out[b * 3 + t] = s + b8[t];
    }
    __syncthreads();
  }
}

// ---------------------------------------------------------------------------
// Host launcher
// ---------------------------------------------------------------------------
extern "C" void kernel_launch(void* const* d_in, const int* in_sizes, int n_in,
                              void* d_out, int out_size, void* d_ws, size_t ws_size,
                              hipStream_t stream) {
  (void)in_sizes; (void)n_in; (void)out_size; (void)ws_size;

  const float* x = (const float*)d_in[0];
  // edge layer l: base 1 + 5l: W, gamma, beta, mean, var
  const float* eW[4];  const float* eG[4]; const float* eB[4];
  const float* eM[4];  const float* eV[4];
  for (int l = 0; l < 4; ++l) {
    eW[l] = (const float*)d_in[1 + 5 * l + 0];
    eG[l] = (const float*)d_in[1 + 5 * l + 1];
    eB[l] = (const float*)d_in[1 + 5 * l + 2];
    eM[l] = (const float*)d_in[1 + 5 * l + 3];
    eV[l] = (const float*)d_in[1 + 5 * l + 4];
  }
  const float* W5  = (const float*)d_in[21];
  const float* g5  = (const float*)d_in[22];
  const float* be5 = (const float*)d_in[23];
  const float* me5 = (const float*)d_in[24];
  const float* va5 = (const float*)d_in[25];
  const float* W6  = (const float*)d_in[26];
  const float* g6  = (const float*)d_in[27];
  const float* be6 = (const float*)d_in[28];
  const float* me6 = (const float*)d_in[29];
  const float* va6 = (const float*)d_in[30];
  const float* W7  = (const float*)d_in[31];
  const float* b7  = (const float*)d_in[32];
  const float* g7  = (const float*)d_in[33];
  const float* be7 = (const float*)d_in[34];
  const float* me7 = (const float*)d_in[35];
  const float* va7 = (const float*)d_in[36];
  const float* W8  = (const float*)d_in[37];
  const float* b8  = (const float*)d_in[38];

  // Workspace carve-up (~37 MB total; fits well inside typical scratch).
  char* ws = (char*)d_ws;
  size_t cur = 0;
  auto alloc = [&](size_t sz) { size_t o = cur; cur = (cur + sz + 255) & ~(size_t)255; return o; };

  __bf16* xt    = (__bf16*)(ws + alloc((size_t)B_ * N_ * 32 * 2));
  __bf16* hcat  = (__bf16*)(ws + alloc((size_t)B_ * N_ * 512 * 2));
  float*  sq    = (float*) (ws + alloc((size_t)B_ * N_ * 4));
  int*    kidx  = (int*)   (ws + alloc((size_t)B_ * N_ * KNN_ * 4));
  __bf16* wb0   = (__bf16*)(ws + alloc((size_t)64 * 64 * 2));
  __bf16* wb1   = (__bf16*)(ws + alloc((size_t)64 * 128 * 2));
  __bf16* wb2   = (__bf16*)(ws + alloc((size_t)128 * 128 * 2));
  __bf16* wb3   = (__bf16*)(ws + alloc((size_t)256 * 256 * 2));
  __bf16* wb5   = (__bf16*)(ws + alloc((size_t)256 * 512 * 2));
  float*  sc0   = (float*) (ws + alloc(64 * 4));
  float*  sh0   = (float*) (ws + alloc(64 * 4));
  float*  sc1   = (float*) (ws + alloc(64 * 4));
  float*  sh1   = (float*) (ws + alloc(64 * 4));
  float*  sc2   = (float*) (ws + alloc(128 * 4));
  float*  sh2   = (float*) (ws + alloc(128 * 4));
  float*  sc3   = (float*) (ws + alloc(256 * 4));
  float*  sh3   = (float*) (ws + alloc(256 * 4));
  float*  sc5   = (float*) (ws + alloc(256 * 4));
  float*  sh5   = (float*) (ws + alloc(256 * 4));
  float*  h5    = (float*) (ws + alloc((size_t)B_ * N_ * 256 * 4));
  float*  gpool = (float*) (ws + alloc((size_t)B_ * 512 * 4));

  const int NP = B_ * N_;
  dim3 thr256(256), thr128(128), thr32(32);

  // Stage 0: bf16 padded point features
  prep_xt<<<(NP + 255) / 256, thr256, 0, stream>>>(x, xt);

  // Weight / BN-fold prep (edge layers + W5)
  prep_weights<<<(64 * 64 + 255) / 256, thr256, 0, stream>>>(
      eW[0], eG[0], eB[0], eM[0], eV[0], wb0, sc0, sh0, 64, 64, 32, 3, 6);
  prep_weights<<<(64 * 128 + 255) / 256, thr256, 0, stream>>>(
      eW[1], eG[1], eB[1], eM[1], eV[1], wb1, sc1, sh1, 64, 128, 64, 64, 128);
  prep_weights<<<(128 * 128 + 255) / 256, thr256, 0, stream>>>(
      eW[2], eG[2], eB[2], eM[2], eV[2], wb2, sc2, sh2, 128, 128, 64, 64, 128);
  prep_weights<<<(256 * 256 + 255) / 256, thr256, 0, stream>>>(
      eW[3], eG[3], eB[3], eM[3], eV[3], wb3, sc3, sh3, 256, 256, 128, 128, 256);
  prep_weights<<<(256 * 512 + 255) / 256, thr256, 0, stream>>>(
      W5, g5, be5, me5, va5, wb5, sc5, sh5, 256, 512, 512, 512, 512);

  // Edge layer 0: feat = xt (stride 32, CPAD 32), out -> hcat[:, 0:64)
  compute_sq<32><<<(NP + 255) / 256, thr256, 0, stream>>>(xt, 32, sq);
  knn_kernel<32><<<NP / 16, thr32, 0, stream>>>(xt, 32, sq, kidx);
  edge_conv<32, 3, 64, 64><<<NP, thr128, 0, stream>>>(
      xt, 32, kidx, wb0, sc0, sh0, hcat + 0, 512);

  // Edge layer 1: feat = hcat[:, 0:64), out -> hcat[:, 64:128)
  compute_sq<64><<<(NP + 255) / 256, thr256, 0, stream>>>(hcat + 0, 512, sq);
  knn_kernel<64><<<NP / 16, thr32, 0, stream>>>(hcat + 0, 512, sq, kidx);
  edge_conv<64, 64, 128, 64><<<NP, thr128, 0, stream>>>(
      hcat + 0, 512, kidx, wb1, sc1, sh1, hcat + 64, 512);

  // Edge layer 2: feat = hcat[:, 64:128), out -> hcat[:, 128:256)
  compute_sq<64><<<(NP + 255) / 256, thr256, 0, stream>>>(hcat + 64, 512, sq);
  knn_kernel<64><<<NP / 16, thr32, 0, stream>>>(hcat + 64, 512, sq, kidx);
  edge_conv<64, 64, 128, 128><<<NP, thr128, 0, stream>>>(
      hcat + 64, 512, kidx, wb2, sc2, sh2, hcat + 128, 512);

  // Edge layer 3: feat = hcat[:, 128:256), out -> hcat[:, 256:512)
  compute_sq<128><<<(NP + 255) / 256, thr256, 0, stream>>>(hcat + 128, 512, sq);
  knn_kernel<128><<<NP / 16, thr32, 0, stream>>>(hcat + 128, 512, sq, kidx);
  edge_conv<128, 128, 256, 256><<<NP, thr128, 0, stream>>>(
      hcat + 128, 512, kidx, wb3, sc3, sh3, hcat + 256, 512);

  // h5 projection + pooling + head
  h5_gemm<<<NP / 16, thr128, 0, stream>>>(hcat, wb5, sc5, sh5, h5);
  pool_kernel<<<B_, thr256, 0, stream>>>(h5, gpool);
  head_kernel<<<1, thr256, 0, stream>>>(gpool, W6, g6, be6, me6, va6,
                                        W7, b7, g7, be7, me7, va7,
                                        W8, b8, (float*)d_out);
}
